// Attention_67559835566462
// MI455X (gfx1250) — matrix operand order repaired and checked by reading, not compiled, not run on hardware
//
#include <hip/hip_runtime.h>

typedef __attribute__((ext_vector_type(16))) _Float16 v16h;
typedef __attribute__((ext_vector_type(8)))  _Float16 v8h;   // 16-byte LDS/global vector
typedef __attribute__((ext_vector_type(2)))  _Float16 h2;    // 4-byte packed pair
typedef __attribute__((ext_vector_type(8)))  float    v8f;
typedef __attribute__((ext_vector_type(4)))  float    f4;

#define WMMA(a,b,c) __builtin_amdgcn_wmma_f32_16x16x32_f16(false,(a),false,(b),(short)0,(c),false,false)

// CDNA5 async global->LDS copy (ASYNCcnt-tracked, no VGPR round trip).
// Generic pointers into __shared__ carry the LDS byte offset in their low 32
// bits (ISA 10.2: LDS aperture keeps addr[31:0]).
#define ASYNC_G2L_B128(ldsPtr, gPtr)                                         \
  asm volatile("global_load_async_to_lds_b128 %0, %1, off"                   \
               :: "v"((unsigned int)(unsigned long long)(ldsPtr)),           \
                  "v"((unsigned long long)(gPtr)) : "memory")
#define ASYNC_WAIT() asm volatile("s_wait_asynccnt 0" ::: "memory")

namespace {
constexpr int SEQ   = 2048;
constexpr int HEADS = 16;
constexpr int DRES  = 1024;
constexpr int DHEAD = 64;
constexpr int MTOT  = 2 * SEQ;   // B*S = 4096 rows
}

__device__ __forceinline__ v8f vzero8() {
  v8f z;
  #pragma unroll
  for (int i = 0; i < 8; ++i) z[i] = 0.0f;
  return z;
}

// 16-half fragment from two 16-byte loads (2x ds_load_b128, no packing).
__device__ __forceinline__ v16h frag16(const _Float16* p0, const _Float16* p1) {
  v8h a = *(const v8h*)p0;
  v8h b = *(const v8h*)p1;
  return __builtin_shufflevector(a, b, 0,1,2,3,4,5,6,7,8,9,10,11,12,13,14,15);
}

// ---------------------------------------------------------------------------
// WMMA GEMM:  C[m,n] = sum_k A[m,k] * W'[k,n] + bias[n]
//   A_F16  : A is f16 workspace (async-copied to LDS) vs f32 input (converted)
//   QKV_W  : W'[k,n] = W[(n>>6)*DRES*DHEAD + k*DHEAD + (n&63)]  else W[k*DRES+n]
//   OUT_F16: f16 workspace output vs f32 final output
// Block 256 threads = 8 waves; tile 64(M) x 128(N); K-step 32.
// ---------------------------------------------------------------------------
template <bool A_F16, bool QKV_W, bool OUT_F16>
__global__ __launch_bounds__(256)
void gemm_wmma(const float* __restrict__ Af32, const _Float16* __restrict__ Af16,
               const float* __restrict__ W0, const float* __restrict__ W1, const float* __restrict__ W2,
               const float* __restrict__ B0, const float* __restrict__ B1, const float* __restrict__ B2,
               _Float16* __restrict__ O0, _Float16* __restrict__ O1, _Float16* __restrict__ O2,
               float* __restrict__ Of)
{
  __shared__ alignas(16) _Float16 sA [64 * 32];    // [m][k] row-major
  __shared__ alignas(16) _Float16 sBt[128 * 32];   // [n][k] k-contiguous

  const int tid  = threadIdx.x;
  const int lane = tid & 31;
  const int wv   = tid >> 5;
  const int m0   = blockIdx.x * 64;
  const int n0   = blockIdx.y * 128;
  const int z    = blockIdx.z;

  const float* W  = (z == 0) ? W0 : (z == 1) ? W1 : W2;
  const float* Bv = (z == 0) ? B0 : (z == 1) ? B1 : B2;
  _Float16*    Oh = (z == 0) ? O0 : (z == 1) ? O1 : O2;

  v8f acc[4];
  #pragma unroll
  for (int i = 0; i < 4; ++i) acc[i] = vzero8();

  const int mA = tid >> 2;           // A: 64 rows, 4 threads/row, 8 halves each
  const int kA = (tid & 3) << 3;
  const int kp  = (tid & 15) << 1;   // B: 2 consecutive k rows ...
  const int nb8 = (tid >> 4) << 3;   //    ... x 8 consecutive n

  const int ml   = lane & 15;
  const int hi   = lane >> 4;
  const int kb8  = hi ? 8 : 0;
  const int kb16 = hi << 4;

  const int wStrideK = QKV_W ? DHEAD : DRES;

  for (int kt = 0; kt < DRES; kt += 32) {
    // ---- stage A ----
    if (A_F16) {
      // pure f16 copy: async DMA straight into LDS
      ASYNC_G2L_B128(&sA[mA * 32 + kA], Af16 + (size_t)(m0 + mA) * DRES + kt + kA);
    } else {
      const f4* s = (const f4*)(Af32 + (size_t)(m0 + mA) * DRES + kt + kA);
      f4 x0 = s[0], x1 = s[1];
      v8h v;
      #pragma unroll
      for (int i = 0; i < 4; ++i) { v[i] = (_Float16)x0[i]; v[4 + i] = (_Float16)x1[i]; }
      *(v8h*)&sA[mA * 32 + kA] = v;
    }
    // ---- stage B transposed (f32 -> f16, k-pairs packed as b32 stores) ----
    {
      const int kg = kt + kp;
      const int n  = n0 + nb8;
      const float* s0 = QKV_W
          ? (W + (size_t)(n >> 6) * (DRES * DHEAD) + (size_t)kg * DHEAD + (n & 63))
          : (W + (size_t)kg * DRES + n);
      const float* s1 = s0 + wStrideK;
      f4 a0 = ((const f4*)s0)[0], a1 = ((const f4*)s0)[1];
      f4 b0 = ((const f4*)s1)[0], b1 = ((const f4*)s1)[1];
      #pragma unroll
      for (int i = 0; i < 4; ++i) {
        h2 p; p[0] = (_Float16)a0[i]; p[1] = (_Float16)b0[i];
        *(h2*)&sBt[(nb8 + i) * 32 + kp] = p;
        h2 q; q[0] = (_Float16)a1[i]; q[1] = (_Float16)b1[i];
        *(h2*)&sBt[(nb8 + 4 + i) * 32 + kp] = q;
      }
      if (kt + 32 < DRES) {
        __builtin_prefetch((const void*)(s0 + (size_t)32 * wStrideK), 0, 1);
        if (!A_F16)
          __builtin_prefetch((const void*)(Af32 + (size_t)(m0 + mA) * DRES + kt + 32 + kA), 0, 1);
      }
    }
    if (A_F16) ASYNC_WAIT();
    __syncthreads();

    // ---- fragments first, then the WMMA chain (overlap LDS latency) ----
    const _Float16* bp = &sBt[((wv << 4) + ml) * 32];
    const v16h bf = frag16(bp + kb8, bp + 16 + kb8);
    v16h af[4];
    #pragma unroll
    for (int i = 0; i < 4; ++i) {
      const _Float16* ap = &sA[((i << 4) + ml) * 32];
      af[i] = frag16(ap + kb8, ap + 16 + kb8);
    }
    #pragma unroll
    for (int i = 0; i < 4; ++i) acc[i] = WMMA(af[i], bf, acc[i]);
    __syncthreads();
  }

  // ---- epilogue: bias + store (coalesced across lanes) ----
  const int ng = n0 + (wv << 4) + ml;
  const float bias = Bv[ng];
  #pragma unroll
  for (int i = 0; i < 4; ++i) {
    #pragma unroll
    for (int r = 0; r < 8; ++r) {
      const int mg = m0 + (i << 4) + r + (hi << 3);
      const float v = acc[i][r] + bias;
      if (OUT_F16) Oh[(size_t)mg * 1024 + ng] = (_Float16)v;
      else         Of[(size_t)mg * 1024 + ng] = v;
    }
  }
}

// ---------------------------------------------------------------------------
// Flash attention per (b,h). Block = 128 threads = 4 waves; 64 query rows per
// block (16 per wave). K tile async-DMA'd to LDS; V staged transposed.
// Wave-uniform causal skipping: blocks fully above the diagonal do no math,
// interior blocks skip the masking VALU.
// ---------------------------------------------------------------------------
__global__ __launch_bounds__(128)
void attn_wmma(const _Float16* __restrict__ Qh, const _Float16* __restrict__ Kh,
               const _Float16* __restrict__ Vh, _Float16* __restrict__ Zb)
{
  __shared__ alignas(16) _Float16 sK [32 * 64];    // [key][c]
  __shared__ alignas(16) _Float16 sVt[64 * 32];    // [c][key]
  __shared__ alignas(16) _Float16 sP [4][16 * 32]; // per-wave P staging

  const int tid  = threadIdx.x;
  const int lane = tid & 31;
  const int wv   = tid >> 5;                    // 0..3
  const int qb   = blockIdx.x * 64;
  const int h    = blockIdx.y;
  const int b    = blockIdx.z;
  const int base = b * SEQ * 1024 + h * 64;     // row stride 1024 (= H*DHEAD)
  const int q0   = qb + (wv << 4);

  const int ml   = lane & 15;
  const int hi   = lane >> 4;
  const int kb8  = hi ? 8 : 0;
  const int kb16 = hi << 4;

  // Q fragments (2 chunks of K=32 over head dim 64) in registers
  v16h aq[2];
  {
    const _Float16* row = Qh + base + (q0 + ml) * 1024;
    #pragma unroll
    for (int c = 0; c < 2; ++c)
      aq[c] = frag16(row + c * 32 + kb8, row + c * 32 + 16 + kb8);
  }

  float Mr[8], Lr[8];
  #pragma unroll
  for (int r = 0; r < 8; ++r) { Mr[r] = -3.0e38f; Lr[r] = 0.0f; }
  v8f o[4];
  #pragma unroll
  for (int c = 0; c < 4; ++c) o[c] = vzero8();

  const int rK = tid >> 2;             // K: 32 rows, 4 threads/row
  const int cK = (tid & 3) << 4;
  const int kpair = (tid & 15) << 1;   // V: 2 consecutive keys ...
  const int cc    = (tid >> 4) << 3;   //    ... x 8 consecutive c

  const int nkb = (qb + 64) >> 5;      // uniform trip count for all 4 waves

  for (int pb = 0; pb < nkb; ++pb) {
    const int p0 = pb << 5;
    {
      // K tile: pure copy -> async DMA to LDS (2 x B128 per thread)
      const _Float16* kr = Kh + base + (p0 + rK) * 1024 + cK;
      ASYNC_G2L_B128(&sK[rK * 64 + cK],     kr);
      ASYNC_G2L_B128(&sK[rK * 64 + cK + 8], kr + 8);

      // V tile: transpose while staging (register path)
      const v8h r0 = *(const v8h*)(Vh + base + (p0 + kpair)     * 1024 + cc);
      const v8h r1 = *(const v8h*)(Vh + base + (p0 + kpair + 1) * 1024 + cc);
      #pragma unroll
      for (int i = 0; i < 8; ++i) {
        h2 p; p[0] = r0[i]; p[1] = r1[i];
        *(h2*)&sVt[(cc + i) * 32 + kpair] = p;
      }
      if (pb + 1 < nkb) {
        __builtin_prefetch((const void*)(Kh + base + (p0 + 32 + rK) * 1024 + cK), 0, 1);
        __builtin_prefetch((const void*)(Vh + base + (p0 + 32 + kpair) * 1024 + cc), 0, 1);
      }
    }
    ASYNC_WAIT();
    __syncthreads();

    // Wave-uniform causal skip: block entirely above this wave's rows -> all
    // scores would be IGNORE, corr==1, rowsum==0: contributes nothing.
    if (p0 <= q0 + 15) {
      // fragments first, then WMMA chain
      v16h kf[2][2];
      #pragma unroll
      for (int nt = 0; nt < 2; ++nt) {
        const int n = (nt << 4) + ml;
        #pragma unroll
        for (int c = 0; c < 2; ++c) {
          const _Float16* bp = &sK[n * 64 + c * 32];
          kf[nt][c] = frag16(bp + kb8, bp + 16 + kb8);
        }
      }
      v8f sc[2];
      sc[0] = vzero8(); sc[1] = vzero8();
      #pragma unroll
      for (int nt = 0; nt < 2; ++nt)
        #pragma unroll
        for (int c = 0; c < 2; ++c)
          sc[nt] = WMMA(aq[c], kf[nt][c], sc[nt]);

      // scale; mask only needed on the diagonal block (wave-uniform test)
      #pragma unroll
      for (int nt = 0; nt < 2; ++nt)
        #pragma unroll
        for (int r = 0; r < 8; ++r) sc[nt][r] *= 0.125f;
      if (p0 + 31 > q0) {
        #pragma unroll
        for (int nt = 0; nt < 2; ++nt) {
          const int kg = p0 + (nt << 4) + ml;
          #pragma unroll
          for (int r = 0; r < 8; ++r) {
            const int qg = q0 + r + (hi << 3);
            if (kg > qg) sc[nt][r] = -100000.0f;
          }
        }
      }

      // online softmax (row spans 16 lanes of same half, both N-tiles)
      #pragma unroll
      for (int r = 0; r < 8; ++r) {
        float mx = fmaxf(sc[0][r], sc[1][r]);
        #pragma unroll
        for (int d = 8; d >= 1; d >>= 1) mx = fmaxf(mx, __shfl_xor(mx, d, 32));
        const float Mn   = fmaxf(Mr[r], mx);
        const float corr = __expf(Mr[r] - Mn);
        const float p0v  = __expf(sc[0][r] - Mn);
        const float p1v  = __expf(sc[1][r] - Mn);
        sc[0][r] = p0v; sc[1][r] = p1v;
        float sum = p0v + p1v;
        #pragma unroll
        for (int d = 8; d >= 1; d >>= 1) sum += __shfl_xor(sum, d, 32);
        Lr[r] = Lr[r] * corr + sum;
        Mr[r] = Mn;
        #pragma unroll
        for (int c = 0; c < 4; ++c) o[c][r] *= corr;
      }

      // stage P: C layout -> LDS -> A-fragment layout (wave-private region)
      _Float16* mp = &sP[wv][0];
      #pragma unroll
      for (int nt = 0; nt < 2; ++nt)
        #pragma unroll
        for (int r = 0; r < 8; ++r)
          mp[(r + (hi << 3)) * 32 + (nt << 4) + ml] = (_Float16)sc[nt][r];
      const _Float16* prow = &mp[ml * 32];
      const v16h pf = frag16(prow + kb8, prow + 16 + kb8);

      v16h vf[4];
      #pragma unroll
      for (int c = 0; c < 4; ++c) {
        const _Float16* vp = &sVt[((c << 4) + ml) * 32];
        vf[c] = frag16(vp + kb8, vp + 16 + kb8);
      }
      #pragma unroll
      for (int c = 0; c < 4; ++c) o[c] = WMMA(pf, vf[c], o[c]);
    }
    __syncthreads();
  }

  // write Z = O / L as f16 [B,S,H*64] (coalesced across lanes)
  #pragma unroll
  for (int c = 0; c < 4; ++c) {
    #pragma unroll
    for (int r = 0; r < 8; ++r) {
      const int mg = q0 + r + (hi << 3);
      Zb[base + mg * 1024 + (c << 4) + ml] = (_Float16)(o[c][r] / Lr[r]);
    }
  }
}

// ---------------------------------------------------------------------------
extern "C" void kernel_launch(void* const* d_in, const int* in_sizes, int n_in,
                              void* d_out, int out_size, void* d_ws, size_t ws_size,
                              hipStream_t stream)
{
  const float* x  = (const float*)d_in[0];
  const float* WQ = (const float*)d_in[1];
  const float* bQ = (const float*)d_in[2];
  const float* WK = (const float*)d_in[3];
  const float* bK = (const float*)d_in[4];
  const float* WV = (const float*)d_in[5];
  const float* bV = (const float*)d_in[6];
  const float* WO = (const float*)d_in[7];
  const float* bO = (const float*)d_in[8];
  float* out = (float*)d_out;

  const size_t plane = (size_t)MTOT * 1024;   // 4M f16 = 8 MB each
  _Float16* Qh = (_Float16*)d_ws;
  _Float16* Kh = Qh + plane;
  _Float16* Vh = Kh + plane;
  _Float16* Zb = Vh + plane;

  // 1) QKV projections (z selects Q/K/V)
  dim3 g1(MTOT / 64, 1024 / 128, 3);
  gemm_wmma<false, true, true><<<g1, 256, 0, stream>>>(
      x, nullptr, WQ, WK, WV, bQ, bK, bV, Qh, Kh, Vh, nullptr);

  // 2) flash attention
  dim3 g2(SEQ / 64, HEADS, 2);
  attn_wmma<<<g2, 128, 0, stream>>>(Qh, Kh, Vh, Zb);

  // 3) output projection + b_O
  dim3 g3(MTOT / 64, 1024 / 128, 1);
  gemm_wmma<true, false, false><<<g3, 256, 0, stream>>>(
      nullptr, Zb, WO, WO, WO, bO, bO, bO, nullptr, nullptr, nullptr, out);
}
